// QuantBasicTransformerBlockINT8_41523743818203
// MI455X (gfx1250) — matrix operand order repaired and checked by reading, not compiled, not run on hardware
//
#include <hip/hip_runtime.h>
#include <math.h>
#include <stdint.h>

// ---------------------------------------------------------------------------
// MI455X (gfx1250) implementation of QuantBasicTransformerBlockINT8.
// GEMM-class math -> v_wmma_f32_16x16x32_f16 (wave32 WMMA, f32 accum).
// B-tile staging -> async global->LDS (ASYNCcnt) in k_gemm and the Tensor
// Data Mover (TENSORcnt) in the GEGLU GEMM, with safe fallbacks.
// ---------------------------------------------------------------------------

typedef __attribute__((ext_vector_type(16))) _Float16 v16h;
typedef __attribute__((ext_vector_type(8)))  float    v8f;
typedef __attribute__((ext_vector_type(4)))  unsigned v4u;
typedef __attribute__((ext_vector_type(8)))  int      v8ii;
typedef __attribute__((ext_vector_type(4)))  int      v4ii;

#define D_MODEL 1024
#define HEADS   16
#define DHEAD   64
#define SEQ     2048
#define BATCH   2
#define CTXLEN  77
#define CTXDIM  768
#define FFDIM   4096
#define LN_EPS  1e-5f

// ---- CDNA5 feature probes (fallbacks keep the build green) ----------------
#if defined(__AMDGCN__) && __has_builtin(__builtin_amdgcn_global_load_async_to_lds_b128)
#define HAS_ASYNC 1
#else
#define HAS_ASYNC 0
#endif
#if defined(__AMDGCN__) && __has_builtin(__builtin_amdgcn_tensor_load_to_lds)
#define HAS_TDM 1
#else
#define HAS_TDM 0
#endif

#define AS1 __attribute__((address_space(1)))
#define AS3 __attribute__((address_space(3)))

// 16B global->LDS copy: async (ASYNCcnt) when available, else load+ds_store.
// Builtin signature (from probe): (v4i AS1* gsrc, v4i AS3* ldst, i32, i32)
static __device__ __forceinline__ void cp_b128(const void* g, void* l) {
#if HAS_ASYNC
  __builtin_amdgcn_global_load_async_to_lds_b128(
      (AS1 v4ii*)(uintptr_t)g, (AS3 v4ii*)(unsigned)(uintptr_t)l, 0, 0);
#else
  *(float4*)l = *(const float4*)g;
#endif
}

static __device__ __forceinline__ void wait_async_copies() {
#if HAS_ASYNC
#if __has_builtin(__builtin_amdgcn_s_wait_asynccnt)
  __builtin_amdgcn_s_wait_asynccnt(0);
#else
  asm volatile("s_wait_asynccnt 0x0" ::: "memory");
#endif
#endif
}

#if HAS_TDM
// Issue a 2D TDM load: tile (tile0 x tile1) of 2-byte elements from a
// row-major tensor with row stride `stride0` (elements) into LDS (row-major,
// tile0 elements per row). D# layout per CDNA5 ISA ch.8.
static __device__ __forceinline__ void tdm_load_2d(unsigned lds_addr,
                                                   const void* gaddr,
                                                   unsigned tdim0, unsigned tdim1,
                                                   unsigned tile0, unsigned tile1,
                                                   unsigned stride0) {
  unsigned long long ga = (unsigned long long)(uintptr_t)gaddr;
  v4u g0;
  g0[0] = 1u;                                   // count=1, user mode
  g0[1] = lds_addr;                             // LDS byte address
  g0[2] = (unsigned)(ga & 0xffffffffu);         // global_addr[31:0]
  g0[3] = (unsigned)((ga >> 32) & 0x01ffffffu)  // global_addr[56:32]
          | (2u << 30);                         // type = 2 ("image")
  v8ii g1;
  g1[0] = (int)(1u << 16);                                  // data_size=2B
  g1[1] = (int)((tdim0 & 0xffffu) << 16);                   // tensor_dim0 lo
  g1[2] = (int)((tdim0 >> 16) | ((tdim1 & 0xffffu) << 16)); // dim0 hi | dim1 lo
  g1[3] = (int)((tdim1 >> 16) | (tile0 << 16));             // dim1 hi | tile0
  g1[4] = (int)(tile1);                                     // tile1, tile2=0
  g1[5] = (int)(stride0);                                   // dim0 stride lo32
  g1[6] = 0;
  g1[7] = 0;
  v4ii z4 = {0, 0, 0, 0};
#if __clang_major__ >= 23
  v8ii z8 = {0, 0, 0, 0, 0, 0, 0, 0};
  __builtin_amdgcn_tensor_load_to_lds(g0, g1, z4, z4, z8, 0);
#else
  __builtin_amdgcn_tensor_load_to_lds(g0, g1, z4, z4, 0);
#endif
}
#endif  // HAS_TDM

static __device__ __forceinline__ v8f wmma_f16(v16h a, v16h b, v8f c) {
  return __builtin_amdgcn_wmma_f32_16x16x32_f16(false, a, false, b, (short)0, c,
                                                false, false);
}

// ---------------------------------------------------------------------------
// Per-tensor min/max reduction (stage 1: 256 blocks of partials)
// ---------------------------------------------------------------------------
__global__ __launch_bounds__(256)
void k_minmax_partial(const float* __restrict__ w, long long n,
                      float* __restrict__ pmin, float* __restrict__ pmax) {
  __shared__ float smn[256], smx[256];
  int tid = threadIdx.x;
  long long i = (long long)blockIdx.x * 256 + tid;
  long long stride = (long long)gridDim.x * 256;
  float mn = 3.0e38f, mx = -3.0e38f;
  for (; i < n; i += stride) { float v = w[i]; mn = fminf(mn, v); mx = fmaxf(mx, v); }
  smn[tid] = mn; smx[tid] = mx; __syncthreads();
  for (int o = 128; o > 0; o >>= 1) {
    if (tid < o) {
      smn[tid] = fminf(smn[tid], smn[tid + o]);
      smx[tid] = fmaxf(smx[tid], smx[tid + o]);
    }
    __syncthreads();
  }
  if (tid == 0) { pmin[blockIdx.x] = smn[0]; pmax[blockIdx.x] = smx[0]; }
}

__global__ __launch_bounds__(256)
void k_minmax_final(const float* __restrict__ pmin, const float* __restrict__ pmax,
                    float* __restrict__ qp) {
  __shared__ float smn[256], smx[256];
  int tid = threadIdx.x;
  smn[tid] = pmin[tid]; smx[tid] = pmax[tid]; __syncthreads();
  for (int o = 128; o > 0; o >>= 1) {
    if (tid < o) {
      smn[tid] = fminf(smn[tid], smn[tid + o]);
      smx[tid] = fmaxf(smx[tid], smx[tid + o]);
    }
    __syncthreads();
  }
  if (tid == 0) {
    float mn = smn[0], mx = smx[0];
    float delta = (mx - mn) / 255.0f;
    float zp = rintf(-mn / delta);
    qp[0] = delta; qp[1] = zp;
  }
}

__global__ __launch_bounds__(256)
void k_quant_f16(const float* __restrict__ w, const float* __restrict__ qp,
                 long long n, _Float16* __restrict__ o) {
  long long i = (long long)blockIdx.x * 256 + threadIdx.x;
  if (i >= n) return;
  float delta = qp[0], zp = qp[1];
  float q = rintf(w[i] / delta) + zp;
  q = fminf(fmaxf(q, 0.0f), 255.0f);
  o[i] = (_Float16)((q - zp) * delta);
}

__global__ __launch_bounds__(256)
void k_cast_f16(const float* __restrict__ x, long long n, _Float16* __restrict__ o) {
  long long i = (long long)blockIdx.x * 256 + threadIdx.x;
  if (i < n) o[i] = (_Float16)x[i];
}

// ---------------------------------------------------------------------------
// LayerNorm over D=1024, one row per block, f16 output.
// ---------------------------------------------------------------------------
__global__ __launch_bounds__(256)
void k_layernorm_f16(const float* __restrict__ x, const float* __restrict__ g,
                     const float* __restrict__ b, _Float16* __restrict__ o) {
  __shared__ float red[256];
  int tid = threadIdx.x;
  const float* xr = x + (size_t)blockIdx.x * D_MODEL;
  float v[4]; float s = 0.f;
#pragma unroll
  for (int i = 0; i < 4; ++i) { v[i] = xr[tid + i * 256]; s += v[i]; }
  red[tid] = s; __syncthreads();
  for (int o2 = 128; o2 > 0; o2 >>= 1) { if (tid < o2) red[tid] += red[tid + o2]; __syncthreads(); }
  float mu = red[0] * (1.0f / D_MODEL);
  __syncthreads();
  s = 0.f;
#pragma unroll
  for (int i = 0; i < 4; ++i) { float d = v[i] - mu; s += d * d; }
  red[tid] = s; __syncthreads();
  for (int o2 = 128; o2 > 0; o2 >>= 1) { if (tid < o2) red[tid] += red[tid + o2]; __syncthreads(); }
  float rstd = rsqrtf(red[0] * (1.0f / D_MODEL) + LN_EPS);
  _Float16* orow = o + (size_t)blockIdx.x * D_MODEL;
#pragma unroll
  for (int i = 0; i < 4; ++i) {
    int c = tid + i * 256;
    orow[c] = (_Float16)((v[i] - mu) * rstd * g[c] + b[c]);
  }
}

// ---------------------------------------------------------------------------
// WMMA GEMM:  Y[M,N] = A[M,K](f16) * W[N,K]^T(f16)
// 8 waves/block; wave = 16 rows x 64 cols; block tile 128x64.
// Double-buffered LDS B tile staged with async global->LDS copies.
// EPI==0: Y -> f16.  EPI==1: outF[row,col] += Y + bias[col].
// ---------------------------------------------------------------------------
template <int EPI>
__global__ __launch_bounds__(256)
void k_gemm(const _Float16* __restrict__ A, const _Float16* __restrict__ W,
            const float* __restrict__ bias, float* __restrict__ outF,
            _Float16* __restrict__ outH, int M, int N, int K) {
  __shared__ __align__(16) _Float16 ldsB[2][64 * 32];
  const int tid = threadIdx.x;
  const int wave = tid >> 5, lane = tid & 31;
  const int l16 = lane & 15, grp = lane >> 4;
  const int mbase = blockIdx.x * 128 + wave * 16;
  const int nbase = blockIdx.y * 64;
  const int m = mbase + l16;
  const int mc = (m < M) ? m : (M - 1);
  const _Float16* Arow = A + (size_t)mc * K;

  const int srow = tid >> 2;          // 0..63: B row within tile
  const int skc  = (tid & 3) << 3;    // 0,8,16,24: k-chunk (halves)
  const _Float16* wsrc = W + (size_t)(nbase + srow) * K + skc;
  _Float16* lds0 = &ldsB[0][srow * 32 + skc];
  _Float16* lds1 = &ldsB[1][srow * 32 + skc];

  v8f z8 = {0.f, 0.f, 0.f, 0.f, 0.f, 0.f, 0.f, 0.f};
  v8f acc[4] = {z8, z8, z8, z8};

  cp_b128(wsrc, lds0);
  wait_async_copies();
  __syncthreads();

  int cur = 0;
  for (int kb = 0; kb < K; kb += 32, cur ^= 1) {
    if (kb + 32 < K) cp_b128(wsrc + kb + 32, cur ? lds0 : lds1);
    // A fragment (16x32 f16, lane half grp holds k = 16*grp + 0..15)
    v16h a;
    {
      const _Float16* p = Arow + kb + grp * 16;
      float4* av = (float4*)&a;
      av[0] = *(const float4*)p;
      av[1] = *(const float4*)(p + 8);
    }
    // all four B fragments first, then the WMMA burst
    v16h bf[4];
#pragma unroll
    for (int s2 = 0; s2 < 4; ++s2) {
      const _Float16* q = &ldsB[cur][(s2 * 16 + l16) * 32 + grp * 16];
      float4* bv = (float4*)&bf[s2];
      bv[0] = *(const float4*)q;
      bv[1] = *(const float4*)(q + 8);
    }
#pragma unroll
    for (int s2 = 0; s2 < 4; ++s2) acc[s2] = wmma_f16(a, bf[s2], acc[s2]);
    wait_async_copies();
    __syncthreads();
  }

#pragma unroll
  for (int s2 = 0; s2 < 4; ++s2) {
    int col = nbase + s2 * 16 + l16;
#pragma unroll
    for (int r = 0; r < 8; ++r) {
      int row = mbase + r + 8 * grp;
      if (row < M) {
        float vv = acc[s2][r];
        if (EPI == 0) outH[(size_t)row * N + col] = (_Float16)vv;
        else          outF[(size_t)row * N + col] += vv + bias[col];
      }
    }
  }
}

// ---------------------------------------------------------------------------
// Fused GEGLU GEMM: h = A*W1[0:FF]^T + b1; g = A*W1[FF:2FF]^T + b1[FF:];
// out = h * gelu_tanh(g) -> f16 [M, FF].  B tiles staged via the TDM
// (tensor_load_to_lds, TENSORcnt) when available.
// ---------------------------------------------------------------------------
__global__ __launch_bounds__(256)
void k_gemm_geglu(const _Float16* __restrict__ A, const _Float16* __restrict__ W,
                  const float* __restrict__ bias, _Float16* __restrict__ outH,
                  int M, int K) {
  __shared__ __align__(16) _Float16 ldsH[2][64 * 32];
  __shared__ __align__(16) _Float16 ldsG[2][64 * 32];
  const int tid = threadIdx.x;
  const int wave = tid >> 5, lane = tid & 31;
  const int l16 = lane & 15, grp = lane >> 4;
  const int mbase = blockIdx.x * 128 + wave * 16;
  const int nbase = blockIdx.y * 64;
  const int m = mbase + l16;
  const int mc = (m < M) ? m : (M - 1);
  const _Float16* Arow = A + (size_t)mc * K;

  const int srow = tid >> 2;
  const int skc  = (tid & 3) << 3;
  const _Float16* wsrcH = W + (size_t)(nbase + srow) * K + skc;
  const _Float16* wsrcG = W + (size_t)(FFDIM + nbase + srow) * K + skc;

  v8f z8 = {0.f, 0.f, 0.f, 0.f, 0.f, 0.f, 0.f, 0.f};
  v8f acch[4] = {z8, z8, z8, z8};
  v8f accg[4] = {z8, z8, z8, z8};

  // ---- staging helpers (TDM preferred, async copy / plain copy fallback) --
#if HAS_TDM
  #define STAGE_GEGLU(buf, kb2)                                                \
    do {                                                                       \
      if (wave == 0) {                                                         \
        tdm_load_2d((unsigned)(uintptr_t)&ldsH[(buf)][0],                      \
                    W + (size_t)nbase * K + (kb2), (unsigned)K,                \
                    (unsigned)(2 * FFDIM), 32u, 64u, (unsigned)K);             \
        tdm_load_2d((unsigned)(uintptr_t)&ldsG[(buf)][0],                      \
                    W + (size_t)(FFDIM + nbase) * K + (kb2), (unsigned)K,      \
                    (unsigned)(2 * FFDIM), 32u, 64u, (unsigned)K);             \
      }                                                                        \
    } while (0)
  #define STAGE_WAIT()                                                         \
    do {                                                                       \
      if (wave == 0) __builtin_amdgcn_s_wait_tensorcnt(0);                     \
      __syncthreads();                                                         \
    } while (0)
#else
  #define STAGE_GEGLU(buf, kb2)                                                \
    do {                                                                       \
      cp_b128(wsrcH + (kb2), &ldsH[(buf)][srow * 32 + skc]);                   \
      cp_b128(wsrcG + (kb2), &ldsG[(buf)][srow * 32 + skc]);                   \
    } while (0)
  #define STAGE_WAIT()                                                         \
    do {                                                                       \
      wait_async_copies();                                                     \
      __syncthreads();                                                         \
    } while (0)
#endif

  STAGE_GEGLU(0, 0);
  STAGE_WAIT();

  int cur = 0;
  for (int kb = 0; kb < K; kb += 32, cur ^= 1) {
    if (kb + 32 < K) STAGE_GEGLU(cur ^ 1, kb + 32);
    v16h a;
    {
      const _Float16* p = Arow + kb + grp * 16;
      float4* av = (float4*)&a;
      av[0] = *(const float4*)p;
      av[1] = *(const float4*)(p + 8);
    }
    v16h bh[4], bg[4];
#pragma unroll
    for (int s2 = 0; s2 < 4; ++s2) {
      const _Float16* qh = &ldsH[cur][(s2 * 16 + l16) * 32 + grp * 16];
      const _Float16* qg = &ldsG[cur][(s2 * 16 + l16) * 32 + grp * 16];
      float4* bhv = (float4*)&bh[s2]; bhv[0] = *(const float4*)qh; bhv[1] = *(const float4*)(qh + 8);
      float4* bgv = (float4*)&bg[s2]; bgv[0] = *(const float4*)qg; bgv[1] = *(const float4*)(qg + 8);
    }
#pragma unroll
    for (int s2 = 0; s2 < 4; ++s2) {
      acch[s2] = wmma_f16(a, bh[s2], acch[s2]);
      accg[s2] = wmma_f16(a, bg[s2], accg[s2]);
    }
    STAGE_WAIT();
  }

#pragma unroll
  for (int s2 = 0; s2 < 4; ++s2) {
    int col = nbase + s2 * 16 + l16;
#pragma unroll
    for (int r = 0; r < 8; ++r) {
      int row = mbase + r + 8 * grp;
      if (row < M) {
        float hv = acch[s2][r] + bias[col];
        float gv = accg[s2][r] + bias[col + FFDIM];
        float gelu = 0.5f * gv *
            (1.0f + tanhf(0.7978845608028654f * (gv + 0.044715f * gv * gv * gv)));
        outH[(size_t)row * FFDIM + col] = (_Float16)(hv * gelu);
      }
    }
  }
#undef STAGE_GEGLU
#undef STAGE_WAIT
}

// ---------------------------------------------------------------------------
// Flash attention (wave32): each wave = 16 query rows, one head.
// ---------------------------------------------------------------------------
__global__ __launch_bounds__(256)
void k_attention(const _Float16* __restrict__ Qm, const _Float16* __restrict__ Km,
                 const _Float16* __restrict__ Vm, _Float16* __restrict__ Om,
                 int qRowsPerB, int kRowsPerB, int Nk) {
  __shared__ __align__(16) _Float16 ldsP[8][16 * 32];
  const int tid = threadIdx.x, wave = tid >> 5, lane = tid & 31;
  const int l16 = lane & 15, grp = lane >> 4;
  const int h = blockIdx.y, bb = blockIdx.z;
  const int qbase = bb * qRowsPerB + blockIdx.x * 128 + wave * 16;
  const int kvbase = bb * kRowsPerB;
  const int hoff = h * DHEAD;
  const int ld = D_MODEL;
  const float scale = 0.125f;  // DH^-0.5

  v16h qa[2];
  {
    const _Float16* p = Qm + (size_t)(qbase + l16) * ld + hoff + grp * 16;
#pragma unroll
    for (int t = 0; t < 2; ++t) {
      float4* av = (float4*)&qa[t];
      av[0] = *(const float4*)(p + t * 32);
      av[1] = *(const float4*)(p + t * 32 + 8);
    }
  }

  v8f z8 = {0.f, 0.f, 0.f, 0.f, 0.f, 0.f, 0.f, 0.f};
  v8f acc[4] = {z8, z8, z8, z8};
  float mr[8], lr[8];
#pragma unroll
  for (int r = 0; r < 8; ++r) { mr[r] = -1e30f; lr[r] = 0.f; }

  for (int jb = 0; jb < Nk; jb += 32) {
    v8f s0 = z8, s1 = z8;
    int key0 = jb + l16;      if (key0 > Nk - 1) key0 = Nk - 1;
    int key1 = jb + 16 + l16; if (key1 > Nk - 1) key1 = Nk - 1;
#pragma unroll
    for (int t = 0; t < 2; ++t) {
      v16h kf0, kf1;
      const _Float16* p0 = Km + (size_t)(kvbase + key0) * ld + hoff + t * 32 + grp * 16;
      const _Float16* p1 = Km + (size_t)(kvbase + key1) * ld + hoff + t * 32 + grp * 16;
      float4* b0 = (float4*)&kf0; b0[0] = *(const float4*)p0; b0[1] = *(const float4*)(p0 + 8);
      float4* b1 = (float4*)&kf1; b1[0] = *(const float4*)p1; b1[1] = *(const float4*)(p1 + 8);
      s0 = wmma_f16(qa[t], kf0, s0);
      s1 = wmma_f16(qa[t], kf1, s1);
    }
    const bool ok0 = (jb + l16) < Nk;
    const bool ok1 = (jb + 16 + l16) < Nk;
#pragma unroll
    for (int r = 0; r < 8; ++r) {
      float v0 = ok0 ? s0[r] * scale : -1e30f;
      float v1 = ok1 ? s1[r] * scale : -1e30f;
      float mx = fmaxf(v0, v1);
#pragma unroll
      for (int dd = 1; dd < 16; dd <<= 1) mx = fmaxf(mx, __shfl_xor(mx, dd, 32));
      float mnew = fmaxf(mr[r], mx);
      float alpha = __expf(mr[r] - mnew);
      float e0 = ok0 ? __expf(v0 - mnew) : 0.f;
      float e1 = ok1 ? __expf(v1 - mnew) : 0.f;
      float rs = e0 + e1;
#pragma unroll
      for (int dd = 1; dd < 16; dd <<= 1) rs += __shfl_xor(rs, dd, 32);
      lr[r] = lr[r] * alpha + rs;
      mr[r] = mnew;
#pragma unroll
      for (int s2 = 0; s2 < 4; ++s2) acc[s2][r] *= alpha;
      int prow = r + 8 * grp;
      ldsP[wave][prow * 32 + l16] = (_Float16)e0;
      ldsP[wave][prow * 32 + 16 + l16] = (_Float16)e1;
    }
    v16h pa;
    {
      const _Float16* p = &ldsP[wave][l16 * 32 + grp * 16];
      float4* av = (float4*)&pa;
      av[0] = *(const float4*)p;
      av[1] = *(const float4*)(p + 8);
    }
#pragma unroll
    for (int s2 = 0; s2 < 4; ++s2) {
      v16h vf;
#pragma unroll
      for (int i = 0; i < 16; ++i) {
        int key = jb + grp * 16 + i; if (key > Nk - 1) key = Nk - 1;
        vf[i] = Vm[(size_t)(kvbase + key) * ld + hoff + s2 * 16 + l16];
      }
      acc[s2] = wmma_f16(pa, vf, acc[s2]);
    }
  }

#pragma unroll
  for (int s2 = 0; s2 < 4; ++s2) {
#pragma unroll
    for (int r = 0; r < 8; ++r) {
      int row = qbase + r + 8 * grp;
      float ov = acc[s2][r] / lr[r];
      Om[(size_t)row * ld + hoff + s2 * 16 + l16] = (_Float16)ov;
    }
  }
}

// ---------------------------------------------------------------------------
// Host orchestration
// ---------------------------------------------------------------------------
extern "C" void kernel_launch(void* const* d_in, const int* in_sizes, int n_in,
                              void* d_out, int out_size, void* d_ws, size_t ws_size,
                              hipStream_t stream) {
  const float* x     = (const float*)d_in[0];
  const float* ctx   = (const float*)d_in[1];
  const float* ln1_g = (const float*)d_in[2];
  const float* ln1_b = (const float*)d_in[3];
  const float* ln2_g = (const float*)d_in[4];
  const float* ln2_b = (const float*)d_in[5];
  const float* ln3_g = (const float*)d_in[6];
  const float* ln3_b = (const float*)d_in[7];
  const float* q1_w = (const float*)d_in[8];
  const float* k1_w = (const float*)d_in[9];
  const float* v1_w = (const float*)d_in[10];
  const float* o1_w = (const float*)d_in[11];
  const float* o1_b = (const float*)d_in[12];
  const float* q2_w = (const float*)d_in[13];
  const float* k2_w = (const float*)d_in[14];
  const float* v2_w = (const float*)d_in[15];
  const float* o2_w = (const float*)d_in[16];
  const float* o2_b = (const float*)d_in[17];
  const float* ff1_w = (const float*)d_in[18];
  const float* ff1_b = (const float*)d_in[19];
  const float* ff2_w = (const float*)d_in[20];
  const float* ff2_b = (const float*)d_in[21];
  float* out = (float*)d_out;

  char* ws = (char*)d_ws;
  size_t off = 0;
  auto alloc = [&](size_t bytes) -> void* {
    void* p = ws + off;
    off += (bytes + 255) & ~(size_t)255;
    return p;
  };

  const int ROWS  = BATCH * SEQ;     // 4096
  const int CROWS = BATCH * CTXLEN;  // 154

  float* qp   = (float*)alloc(10 * 2 * sizeof(float));
  float* pmin = (float*)alloc(256 * sizeof(float));
  float* pmax = (float*)alloc(256 * sizeof(float));

  _Float16* wq1 = (_Float16*)alloc((size_t)D_MODEL * D_MODEL * 2);
  _Float16* wk1 = (_Float16*)alloc((size_t)D_MODEL * D_MODEL * 2);
  _Float16* wv1 = (_Float16*)alloc((size_t)D_MODEL * D_MODEL * 2);
  _Float16* wo1 = (_Float16*)alloc((size_t)D_MODEL * D_MODEL * 2);
  _Float16* wq2 = (_Float16*)alloc((size_t)D_MODEL * D_MODEL * 2);
  _Float16* wk2 = (_Float16*)alloc((size_t)D_MODEL * CTXDIM * 2);
  _Float16* wv2 = (_Float16*)alloc((size_t)D_MODEL * CTXDIM * 2);
  _Float16* wo2 = (_Float16*)alloc((size_t)D_MODEL * D_MODEL * 2);
  _Float16* wf1 = (_Float16*)alloc((size_t)2 * FFDIM * D_MODEL * 2);
  _Float16* wf2 = (_Float16*)alloc((size_t)D_MODEL * FFDIM * 2);

  _Float16* xln  = (_Float16*)alloc((size_t)ROWS * D_MODEL * 2);
  _Float16* qbuf = (_Float16*)alloc((size_t)ROWS * D_MODEL * 2);
  _Float16* kbuf = (_Float16*)alloc((size_t)ROWS * D_MODEL * 2);
  _Float16* vbuf = (_Float16*)alloc((size_t)ROWS * D_MODEL * 2);
  _Float16* aout = (_Float16*)alloc((size_t)ROWS * D_MODEL * 2);
  _Float16* c16  = (_Float16*)alloc((size_t)CROWS * CTXDIM * 2);
  _Float16* h16  = (_Float16*)alloc((size_t)ROWS * FFDIM * 2);

  struct { const float* w; long long n; _Float16* o; } WL[10] = {
    {q1_w, (long long)D_MODEL * D_MODEL, wq1},
    {k1_w, (long long)D_MODEL * D_MODEL, wk1},
    {v1_w, (long long)D_MODEL * D_MODEL, wv1},
    {o1_w, (long long)D_MODEL * D_MODEL, wo1},
    {q2_w, (long long)D_MODEL * D_MODEL, wq2},
    {k2_w, (long long)D_MODEL * CTXDIM,  wk2},
    {v2_w, (long long)D_MODEL * CTXDIM,  wv2},
    {o2_w, (long long)D_MODEL * D_MODEL, wo2},
    {ff1_w, (long long)2 * FFDIM * D_MODEL, wf1},
    {ff2_w, (long long)D_MODEL * FFDIM,  wf2},
  };
  for (int t = 0; t < 10; ++t) {
    k_minmax_partial<<<256, 256, 0, stream>>>(WL[t].w, WL[t].n, pmin, pmax);
    k_minmax_final<<<1, 256, 0, stream>>>(pmin, pmax, qp + 2 * t);
    int blocks = (int)((WL[t].n + 255) / 256);
    k_quant_f16<<<blocks, 256, 0, stream>>>(WL[t].w, qp + 2 * t, WL[t].n, WL[t].o);
  }

  (void)hipMemcpyAsync(out, x, (size_t)out_size * sizeof(float),
                       hipMemcpyDeviceToDevice, stream);

  auto gemm16 = [&](const _Float16* A, const _Float16* W, _Float16* Y,
                    int M, int N, int K) {
    dim3 grid((M + 127) / 128, N / 64);
    k_gemm<0><<<grid, 256, 0, stream>>>(A, W, nullptr, nullptr, Y, M, N, K);
  };
  auto gemmacc = [&](const _Float16* A, const _Float16* W, const float* bias,
                     float* Y, int M, int N, int K) {
    dim3 grid((M + 127) / 128, N / 64);
    k_gemm<1><<<grid, 256, 0, stream>>>(A, W, bias, Y, nullptr, M, N, K);
  };

  // ---- sublayer 1: self attention ----
  k_layernorm_f16<<<ROWS, 256, 0, stream>>>(out, ln1_g, ln1_b, xln);
  gemm16(xln, wq1, qbuf, ROWS, D_MODEL, D_MODEL);
  gemm16(xln, wk1, kbuf, ROWS, D_MODEL, D_MODEL);
  gemm16(xln, wv1, vbuf, ROWS, D_MODEL, D_MODEL);
  k_attention<<<dim3(SEQ / 128, HEADS, BATCH), 256, 0, stream>>>(
      qbuf, kbuf, vbuf, aout, SEQ, SEQ, SEQ);
  gemmacc(aout, wo1, o1_b, out, ROWS, D_MODEL, D_MODEL);

  // ---- sublayer 2: cross attention ----
  k_layernorm_f16<<<ROWS, 256, 0, stream>>>(out, ln2_g, ln2_b, xln);
  {
    long long nctx = (long long)CROWS * CTXDIM;
    k_cast_f16<<<(int)((nctx + 255) / 256), 256, 0, stream>>>(ctx, nctx, c16);
  }
  gemm16(xln, wq2, qbuf, ROWS, D_MODEL, D_MODEL);
  gemm16(c16, wk2, kbuf, CROWS, D_MODEL, CTXDIM);
  gemm16(c16, wv2, vbuf, CROWS, D_MODEL, CTXDIM);
  k_attention<<<dim3(SEQ / 128, HEADS, BATCH), 256, 0, stream>>>(
      qbuf, kbuf, vbuf, aout, SEQ, CTXLEN, CTXLEN);
  gemmacc(aout, wo2, o2_b, out, ROWS, D_MODEL, D_MODEL);

  // ---- sublayer 3: GEGLU feed-forward ----
  k_layernorm_f16<<<ROWS, 256, 0, stream>>>(out, ln3_g, ln3_b, xln);
  k_gemm_geglu<<<dim3(ROWS / 128, FFDIM / 64), 256, 0, stream>>>(
      xln, wf1, ff1_b, h16, ROWS, D_MODEL);
  gemmacc(h16, wf2, ff2_b, out, ROWS, D_MODEL, FFDIM);
}
